// SecConv2d_2628519985886
// MI455X (gfx1250) — 8-bit matrix operand order repaired and checked by reading, compiled, not run on hardware
//
#include <hip/hip_runtime.h>


typedef __attribute__((ext_vector_type(8))) int v8i;

namespace {
constexpr int C       = 64;
constexpr int KOUT    = 64;
constexpr int H       = 56;
constexpr int W       = 56;
constexpr int NIMG    = 32;
constexpr int OH_TILE = 4;
constexpr int XROWS   = OH_TILE + 2;
constexpr int XW      = W + 2;
constexpr int STRIPS  = H / OH_TILE;
constexpr int NTILES  = (OH_TILE * W) / 16;
constexpr int WBYTES  = KOUT * 9 * C;
constexpr int XBYTES  = XROWS * XW * C;
constexpr int SMEM_BYTES = 2 * WBYTES + 2 * XBYTES + KOUT * 8;
}

__device__ __forceinline__ v8i load_a_frag(const unsigned char* p, int ab) {
  uint2 a0 = *(const uint2*)(p + ab +  0);
  uint2 a1 = *(const uint2*)(p + ab + 16);
  uint2 a2 = *(const uint2*)(p + ab + 32);
  uint2 a3 = *(const uint2*)(p + ab + 48);
  v8i r;
  r[0] = (int)a0.x; r[1] = (int)a0.y; r[2] = (int)a1.x; r[3] = (int)a1.y;
  r[4] = (int)a2.x; r[5] = (int)a2.y; r[6] = (int)a3.x; r[7] = (int)a3.y;
  return r;
}

__device__ __forceinline__ v8i load_b_frag(const unsigned char* p, int bb) {
  int4 b0 = *(const int4*)(p + bb +  0);
  int4 b1 = *(const int4*)(p + bb + 32);
  v8i r;
  r[0] = b0.x; r[1] = b0.y; r[2] = b0.z; r[3] = b0.w;
  r[4] = b1.x; r[5] = b1.y; r[6] = b1.z; r[7] = b1.w;
  return r;
}

__global__ __launch_bounds__(256) void secconv2d_wmma_iu8(
    const unsigned long long* __restrict__ x,
    const unsigned long long* __restrict__ w,
    const unsigned long long* __restrict__ bias,
    unsigned long long* __restrict__ out)
{
  extern __shared__ unsigned char smem[];
  unsigned char* wlo = smem;
  unsigned char* whi = wlo + WBYTES;
  unsigned char* xlo = whi + WBYTES;
  unsigned char* xhi = xlo + XBYTES;
  unsigned long long* blds = (unsigned long long*)(xhi + XBYTES);

  const int tid = threadIdx.x;
  const int n   = blockIdx.x / STRIPS;
  const int r0  = (blockIdx.x % STRIPS) * OH_TILE;

  for (int g = tid; g < KOUT * 9 * (C / 4); g += 256) {
    const int kout = g / (9 * (C / 4));
    const int rem  = g - kout * (9 * (C / 4));
    const int t    = rem >> 4;
    const int c0   = (rem & 15) << 2;
    unsigned lo = 0, hi = 0;
#pragma unroll
    for (int j = 0; j < 4; ++j) {
      const unsigned u = (unsigned)w[(kout * C + c0 + j) * 9 + t];
      lo |= (u & 0xFFu) << (8 * j);
      hi |= ((u >> 8) & 0xFFu) << (8 * j);
    }
    const int o = (kout * 9 + t) * C + c0;
    *(unsigned*)(wlo + o) = lo;
    *(unsigned*)(whi + o) = hi;
  }
  if (tid < KOUT) blds[tid] = bias[tid];

  const unsigned long long* xn = x + (size_t)n * C * H * W;
  for (int g = tid; g < (C / 4) * XROWS * XW; g += 256) {
    const int cg  = g / (XROWS * XW);
    const int rem = g - cg * (XROWS * XW);
    const int yy  = rem / XW;
    const int xx  = rem - yy * XW;
    const int gy  = r0 + yy - 1;
    const int gx  = xx - 1;
    const int c0  = cg << 2;
    unsigned lo = 0, hi = 0;
    if (gy >= 0 && gy < H && gx >= 0 && gx < W) {
#pragma unroll
      for (int j = 0; j < 4; ++j) {
        const unsigned u = (unsigned)xn[((c0 + j) * H + gy) * W + gx];
        lo |= (u & 0xFFu) << (8 * j);
        hi |= ((u >> 8) & 0xFFu) << (8 * j);
      }
    }
    const int o = (yy * XW + xx) * C + c0;
    *(unsigned*)(xlo + o) = lo;
    *(unsigned*)(xhi + o) = hi;
  }
  __syncthreads();

  const int wave = tid >> 5;
  const int lane = tid & 31;
  const int half = lane >> 4;
  const int lq   = lane & 15;

  for (int tile = wave; tile < 4 * NTILES; tile += 8) {
    const int mt = tile / NTILES;
    const int nt = tile - mt * NTILES;
    const int l  = nt * 16 + lq;
    const int oy = l / W;
    const int ox = l - oy * W;
    const int spbase = (oy * XW + ox) * C;
    const int abase0 = (mt * 16 + lq) * 9 * C + half * 16;

    v8i acc_ll = {0,0,0,0,0,0,0,0};
    v8i acc_md = {0,0,0,0,0,0,0,0};
    v8i acc_hh = {0,0,0,0,0,0,0,0};

#pragma unroll
    for (int t = 0; t < 9; ++t) {
      const int di = t / 3;
      const int dj = t - di * 3;
      const int ab = abase0 + t * C;
      const int bb = spbase + (di * XW + dj) * C + half * 16;

      const v8i A_lo = load_b_frag(wlo, ab);
      const v8i A_hi = load_b_frag(whi, ab);
      const v8i B_lo = load_b_frag(xlo, bb);
      const v8i B_hi = load_b_frag(xhi, bb);

      acc_ll = __builtin_amdgcn_wmma_i32_16x16x64_iu8(false, A_lo, false, B_lo, acc_ll, false, false);
      acc_md = __builtin_amdgcn_wmma_i32_16x16x64_iu8(false, A_lo, false, B_hi, acc_md, false, false);
      acc_md = __builtin_amdgcn_wmma_i32_16x16x64_iu8(false, A_hi, false, B_lo, acc_md, false, false);
      acc_hh = __builtin_amdgcn_wmma_i32_16x16x64_iu8(false, A_hi, false, B_hi, acc_hh, false, false);
    }

    const int gy = r0 + oy;
    const int m0 = mt * 16 + half * 8;
    unsigned long long* outp =
        out + (((size_t)n * KOUT + m0) * H + gy) * W + ox;
#pragma unroll
    for (int v = 0; v < 8; ++v) {
      const unsigned long long r =
          (unsigned long long)(unsigned)acc_ll[v] +
          ((unsigned long long)(unsigned)acc_md[v] << 8) +
          ((unsigned long long)(unsigned)acc_hh[v] << 16) +
          blds[m0 + v];
      outp[(size_t)v * H * W] = r;
    }
  }
}

extern "C" void kernel_launch(void* const* d_in, const int* in_sizes, int n_in,
                              void* d_out, int out_size, void* d_ws, size_t ws_size,
                              hipStream_t stream) {
  (void)in_sizes; (void)n_in; (void)out_size; (void)d_ws; (void)ws_size;
  const unsigned long long* x  = (const unsigned long long*)d_in[0];
  const unsigned long long* wt = (const unsigned long long*)d_in[1];
  const unsigned long long* b  = (const unsigned long long*)d_in[2];
  unsigned long long* out = (unsigned long long*)d_out;

  dim3 grid(NIMG * STRIPS);
  dim3 block(256);
  secconv2d_wmma_iu8<<<grid, block, SMEM_BYTES, stream>>>(x, wt, b, out);
}
